// dgl_GNNModel_77086073028873
// MI455X (gfx1250) — matrix operand order repaired and checked by reading, not compiled, not run on hardware
//
#include <hip/hip_runtime.h>
#include <hip/hip_bf16.h>

// ---------------------------------------------------------------------------
// GAT (3-layer) for MI455X / gfx1250.
//   N=100000 nodes, E=1600000 edges, feature widths 256->128->128->128.
//   GEMMs via v_wmma_f32_16x16x32_bf16 (wave32; 2x 16x16 C tiles per wave,
//   B fragment reused across both -> halves packed-W L2 traffic).
//   Edge softmax/aggregation via L2-resident gathers + f32 global atomics.
// ---------------------------------------------------------------------------

#define NN 100000
#define EE 1600000

typedef __attribute__((ext_vector_type(16))) __bf16 v16bf;
typedef __attribute__((ext_vector_type(8)))  float  v8f;

__device__ __forceinline__ unsigned short f32_to_bf16_bits(float v) {
  unsigned u = __float_as_uint(v);
  unsigned r = (u + 0x7FFFu + ((u >> 16) & 1u)) >> 16;   // round-to-nearest-even
  return (unsigned short)r;
}
__device__ __forceinline__ __bf16 bf16_from_bits(unsigned short s) {
  union { unsigned short u; __bf16 b; } x; x.u = s; return x.b;
}
// monotonic float<->uint mapping for atomicMax-based segment max
__device__ __forceinline__ unsigned fkey(float f) {
  unsigned u = __float_as_uint(f);
  return u ^ ((unsigned)(((int)u) >> 31) | 0x80000000u);
}
__device__ __forceinline__ float funkey(unsigned k) {
  unsigned u = (k & 0x80000000u) ? (k ^ 0x80000000u) : ~k;
  return __uint_as_float(u);
}
#define FKEY_NEG_INF 0x007FFFFFu   // fkey(-inf)

// ---------------------------------------------------------------------------
// Pack W (K x 128, f32 row-major) into per-wave B fragments (bf16):
//   element idx = ((kt*8 + ct)*32 + lane)*16 + e
//   K = kt*32 + (e>=8 ? 16 : 0) + (lane>=16 ? 8 : 0) + (e&7)
//   col     = ct*16 + (lane&15)
// ---------------------------------------------------------------------------
__global__ void gat_pack_w(const float* __restrict__ W,
                           unsigned short* __restrict__ Wp, int K) {
  int idx = blockIdx.x * 256 + threadIdx.x;
  if (idx >= K * 128) return;
  int e  = idx & 15;
  int L  = (idx >> 4) & 31;
  int ct = (idx >> 9) & 7;
  int kt = idx >> 12;
  int k   = kt * 32 + ((e >= 8) ? 16 : 0) + ((L >= 16) ? 8 : 0) + (e & 7);
  int col = ct * 16 + (L & 15);
  Wp[idx] = f32_to_bf16_bits(W[k * 128 + col]);
}

// ---------------------------------------------------------------------------
// feat(N x 128) = X(N x K) @ W(K x 128), bf16 WMMA, f32 accumulate.
// grid = N/32 row tiles (32 rows/block), 256 threads = 8 waves;
// wave w -> column tile w; each wave computes two stacked 16x16 C tiles,
// reusing each B fragment for both WMMAs.
// ---------------------------------------------------------------------------
__global__ void gat_gemm_wmma(const float* __restrict__ X,
                              const unsigned short* __restrict__ Wp,
                              float* __restrict__ Y, int K, int kbits) {
  extern __shared__ unsigned short Alds[];   // [32][KP] bf16 bits
  const int KP  = K + 8;                     // pad rows (16B) to spread LDS banks
  const int bt  = blockIdx.x;
  const int tid = threadIdx.x;
  const int w   = tid >> 5;                  // wave id == column tile
  const int L   = tid & 31;

  // stage A tile (32 x K) into LDS as bf16, coalesced along K
  const float* Ag = X + (size_t)bt * 32 * K;
  const int elems = 32 * K;
  for (int i = tid; i < elems; i += 256) {
    int row = i >> kbits;
    int k   = i & (K - 1);
    Alds[row * KP + k] = f32_to_bf16_bits(Ag[i]);
  }
  __syncthreads();

  v8f c0 = {0.f, 0.f, 0.f, 0.f, 0.f, 0.f, 0.f, 0.f};
  v8f c1 = {0.f, 0.f, 0.f, 0.f, 0.f, 0.f, 0.f, 0.f};
  const int nkt = K >> 5;
  const unsigned short* rowp0 =
      &Alds[(L & 15) * KP + ((L >= 16) ? 8 : 0)];          // rows 0..15
  const unsigned short* rowp1 = rowp0 + 16 * KP;           // rows 16..31

  for (int kt = 0; kt < nkt; ++kt) {
    // A fragments (16-bit A layout, ISA 7.12.2): k = e + (e>=8?8:0) + (L>=16?8:0)
    const unsigned short* p0 = rowp0 + kt * 32;
    const unsigned short* p1 = rowp1 + kt * 32;
    v16bf a0, a1;
#pragma unroll
    for (int e = 0; e < 8; ++e) {
      a0[e]     = bf16_from_bits(p0[e]);
      a0[e + 8] = bf16_from_bits(p0[e + 16]);
      a1[e]     = bf16_from_bits(p1[e]);
      a1[e + 8] = bf16_from_bits(p1[e + 16]);
    }
    // B fragment: one contiguous 32-byte load per lane, reused for both tiles
    v16bf b = *reinterpret_cast<const v16bf*>(
        Wp + ((size_t)(kt * 8 + w) * 32 + L) * 16);
    c0 = __builtin_amdgcn_wmma_f32_16x16x32_bf16(
        false, a0, false, b, (short)0, c0, false, false);
    c1 = __builtin_amdgcn_wmma_f32_16x16x32_bf16(
        false, a1, false, b, (short)0, c1, false, false);
  }

  // C layout: VGPR i -> M = i + (lane>=16 ? 8 : 0), Ncol = lane & 15
  const int col   = (w << 4) + (L & 15);
  const int rbase = bt * 32 + ((L >= 16) ? 8 : 0);
#pragma unroll
  for (int i = 0; i < 8; ++i) {
    Y[(size_t)(rbase + i) * 128 + col]      = c0[i];
    Y[(size_t)(rbase + 16 + i) * 128 + col] = c1[i];
  }
}

// ---------------------------------------------------------------------------
// el/er: wave per node; lane handles 4 contiguous columns; head-group reduce.
// ---------------------------------------------------------------------------
__global__ void gat_elr(const float* __restrict__ feat,
                        const float* __restrict__ al,
                        const float* __restrict__ ar,
                        float* __restrict__ el, float* __restrict__ er, int H) {
  int wid = (blockIdx.x * 256 + threadIdx.x) >> 5;
  int L   = threadIdx.x & 31;
  if (wid >= NN) return;
  float4 f = reinterpret_cast<const float4*>(feat + (size_t)wid * 128)[L];
  float4 a = reinterpret_cast<const float4*>(al)[L];
  float4 r = reinterpret_cast<const float4*>(ar)[L];
  float pl = f.x * a.x + f.y * a.y + f.z * a.z + f.w * a.w;
  float pr = f.x * r.x + f.y * r.y + f.z * r.z + f.w * r.w;
  int G = 32 / H;                              // lanes per head
  for (int off = G >> 1; off > 0; off >>= 1) {
    pl += __shfl_xor(pl, off, 32);
    pr += __shfl_xor(pr, off, 32);
  }
  if ((L & (G - 1)) == 0) {
    int h = L / G;
    el[(size_t)wid * H + h] = pl;
    er[(size_t)wid * H + h] = pr;
  }
}

__global__ void gat_init_u32(unsigned* __restrict__ p, unsigned v, int n) {
  int i = blockIdx.x * 256 + threadIdx.x;
  if (i < n) p[i] = v;
}

// pass A: segment max of leaky(el[src]+er[dst]) over dst
__global__ void gat_edge_max(const int* __restrict__ src, const int* __restrict__ dst,
                             const float* __restrict__ el, const float* __restrict__ er,
                             unsigned* __restrict__ mkey, int H, int hshift, int EH) {
  int idx = blockIdx.x * 256 + threadIdx.x;
  if (idx >= EH) return;
  int e = idx >> hshift, h = idx & (H - 1);
  int s = src[e], d = dst[e];
  float t  = el[(size_t)s * H + h] + er[(size_t)d * H + h];
  float lk = (t > 0.f) ? t : 0.2f * t;
  atomicMax(&mkey[(size_t)d * H + h], fkey(lk));
}

// pass B: segment sum of exp(e - m[dst])
__global__ void gat_edge_expsum(const int* __restrict__ src, const int* __restrict__ dst,
                                const float* __restrict__ el, const float* __restrict__ er,
                                const unsigned* __restrict__ mkey,
                                float* __restrict__ ssum, int H, int hshift, int EH) {
  int idx = blockIdx.x * 256 + threadIdx.x;
  if (idx >= EH) return;
  int e = idx >> hshift, h = idx & (H - 1);
  int s = src[e], d = dst[e];
  float t  = el[(size_t)s * H + h] + er[(size_t)d * H + h];
  float lk = (t > 0.f) ? t : 0.2f * t;
  float ex = expf(lk - funkey(mkey[(size_t)d * H + h]));
  atomicAdd(&ssum[(size_t)d * H + h], ex);
}

// pass C: out[dst] += alpha * feat[src]; wave per edge, float4 per lane
__global__ void gat_edge_aggr(const int* __restrict__ src, const int* __restrict__ dst,
                              const float* __restrict__ el, const float* __restrict__ er,
                              const unsigned* __restrict__ mkey,
                              const float* __restrict__ ssum,
                              const float* __restrict__ feat,
                              float* __restrict__ out, int H, int dshift) {
  int gw = (blockIdx.x * 256 + threadIdx.x) >> 5;
  int L  = threadIdx.x & 31;
  if (gw >= EE) return;
  int s = src[gw], d = dst[gw];
  int col = L << 2;
  int h   = col >> dshift;                     // head of this lane's columns
  float t  = el[(size_t)s * H + h] + er[(size_t)d * H + h];
  float lk = (t > 0.f) ? t : 0.2f * t;
  float alpha = expf(lk - funkey(mkey[(size_t)d * H + h])) / ssum[(size_t)d * H + h];
  float4 f = reinterpret_cast<const float4*>(feat + (size_t)s * 128)[L];
  float* op = out + (size_t)d * 128 + col;
  atomicAdd(op + 0, alpha * f.x);
  atomicAdd(op + 1, alpha * f.y);
  atomicAdd(op + 2, alpha * f.z);
  atomicAdd(op + 3, alpha * f.w);
}

// bias + optional ELU, in place
__global__ void gat_finalize(float* __restrict__ Y, const float* __restrict__ b,
                             int elu, int total) {
  int idx = blockIdx.x * 256 + threadIdx.x;
  if (idx >= total) return;
  float v = Y[idx] + b[idx & 127];
  if (elu) v = (v > 0.f) ? v : expm1f(v);
  Y[idx] = v;
}

// ---------------------------------------------------------------------------
namespace {
void run_layer(const float* X, int K, int kbits,
               const float* W, const float* al, const float* ar, const float* b,
               int H, int hshift, int dshift, int elu,
               const int* src, const int* dst,
               float* featW, float* Y, unsigned short* wpack,
               float* elb, float* erb, unsigned* mk, float* ss,
               hipStream_t stream) {
  gat_pack_w<<<dim3((K * 128 + 255) / 256), dim3(256), 0, stream>>>(W, wpack, K);
  size_t lds = (size_t)32 * (K + 8) * sizeof(unsigned short);
  gat_gemm_wmma<<<dim3(NN / 32), dim3(256), lds, stream>>>(X, wpack, featW, K, kbits);
  gat_elr<<<dim3(NN * 32 / 256), dim3(256), 0, stream>>>(featW, al, ar, elb, erb, H);

  int nh = NN * H;
  gat_init_u32<<<dim3((nh + 255) / 256), dim3(256), 0, stream>>>(mk, FKEY_NEG_INF, nh);
  hipMemsetAsync(ss, 0, (size_t)nh * sizeof(float), stream);

  int EH = EE * H;
  gat_edge_max<<<dim3((EH + 255) / 256), dim3(256), 0, stream>>>(
      src, dst, elb, erb, mk, H, hshift, EH);
  gat_edge_expsum<<<dim3((EH + 255) / 256), dim3(256), 0, stream>>>(
      src, dst, elb, erb, mk, ss, H, hshift, EH);

  hipMemsetAsync(Y, 0, (size_t)NN * 128 * sizeof(float), stream);
  gat_edge_aggr<<<dim3(EE * 32 / 256), dim3(256), 0, stream>>>(
      src, dst, elb, erb, mk, ss, featW, Y, H, dshift);
  gat_finalize<<<dim3(NN * 128 / 256), dim3(256), 0, stream>>>(
      Y, b, elu, NN * 128);
}
} // namespace

extern "C" void kernel_launch(void* const* d_in, const int* in_sizes, int n_in,
                              void* d_out, int out_size, void* d_ws, size_t ws_size,
                              hipStream_t stream) {
  const float* feat = (const float*)d_in[0];
  const int*   src  = (const int*)d_in[1];
  const int*   dst  = (const int*)d_in[2];
  const float* W0 = (const float*)d_in[3],  *al0 = (const float*)d_in[4];
  const float* ar0 = (const float*)d_in[5], *b0  = (const float*)d_in[6];
  const float* W1 = (const float*)d_in[7],  *al1 = (const float*)d_in[8];
  const float* ar1 = (const float*)d_in[9], *b1  = (const float*)d_in[10];
  const float* W2 = (const float*)d_in[11], *al2 = (const float*)d_in[12];
  const float* ar2 = (const float*)d_in[13],*b2  = (const float*)d_in[14];
  float* out = (float*)d_out;

  // workspace carve-up
  char* ws = (char*)d_ws;
  const size_t NB = (size_t)NN * 128 * sizeof(float);       // 51.2 MB
  float*          featW = (float*)ws;                       // GEMM output
  float*          bufB  = (float*)(ws + NB);                // layer activations
  float*          elb   = (float*)(ws + 2 * NB);            // N*8
  float*          erb   = elb + (size_t)NN * 8;
  unsigned*       mk    = (unsigned*)(erb + (size_t)NN * 8);
  float*          ss    = (float*)(mk + (size_t)NN * 8);
  unsigned short* wpack = (unsigned short*)(ss + (size_t)NN * 8);  // <=64 KB

  // layer 0: 256 -> 128, H=1, ELU
  run_layer(feat, 256, 8, W0, al0, ar0, b0, 1, 0, 7, 1,
            src, dst, featW, bufB, wpack, elb, erb, mk, ss, stream);
  // layer 1: 128 -> 128, H=1, ELU (head-mean over 1 head == identity)
  run_layer(bufB, 128, 7, W1, al1, ar1, b1, 1, 0, 7, 1,
            src, dst, featW, bufB, wpack, elb, erb, mk, ss, stream);
  // layer 2: 128 -> 8 heads x 16, no activation, straight into d_out
  run_layer(bufB, 128, 7, W2, al2, ar2, b2, 8, 3, 4, 0,
            src, dst, featW, out, wpack, elb, erb, mk, ss, stream);
}
